// Fuser_30374008717427
// MI455X (gfx1250) — matrix operand order repaired and checked by reading, not compiled, not run on hardware
//
#include <hip/hip_runtime.h>
#include <hip/hip_bf16.h>

// ---------------------------------------------------------------------------
// Problem constants (match reference)
// ---------------------------------------------------------------------------
#define H_    8
#define D_    1024
#define DH_   128
#define KDIM_ 3072
#define NS_   2048
#define NK_   8192

typedef __attribute__((ext_vector_type(16))) __bf16        v16bf;
typedef __attribute__((ext_vector_type(8)))  float         v8f;
typedef __attribute__((ext_vector_type(4)))  unsigned int  u32x4;
typedef __attribute__((ext_vector_type(8)))  int           i32x8;
typedef __attribute__((ext_vector_type(4)))  int           i32x4;

// TDM availability (probe: builtin exists on both toolchains, arity differs)
#if __has_builtin(__builtin_amdgcn_tensor_load_to_lds) && \
    __has_builtin(__builtin_amdgcn_s_wait_tensorcnt)
#define USE_TDM 1
#if __clang_major__ >= 23
#define TDM_6ARG 1
#endif
#endif

// f32 -> bf16, round-to-nearest-even
__device__ __forceinline__ unsigned short f2bf(float f) {
  unsigned int u = __float_as_uint(f);
  u += 0x7FFFu + ((u >> 16) & 1u);
  return (unsigned short)(u >> 16);
}
// pack two f32 -> 2xbf16 in one dword (lo in [15:0])
__device__ __forceinline__ unsigned int pk2(float lo, float hi) {
  return (unsigned int)f2bf(lo) | ((unsigned int)f2bf(hi) << 16);
}

union Frag16 { uint4 q[2]; v16bf v; };

// ---------------------------------------------------------------------------
// TDM: async 2D bf16 tile copy global -> LDS (tileW elems x tileH rows,
// global row stride = rowStride elems, LDS rows stored contiguously).
// D# bitfields per CDNA5 ISA 8.3-8.6. Issue from one wave; EXEC ignored.
// ---------------------------------------------------------------------------
#ifdef USE_TDM
__device__ __forceinline__ void tdm_load_2d_bf16(const unsigned short* gtile,
                                                 unsigned int ldsOff,
                                                 unsigned int tileW,
                                                 unsigned int tileH,
                                                 unsigned int rowStride)
{
  unsigned long long ga = (unsigned long long)(size_t)gtile;
  u32x4 g0;
  g0[0] = 1u;                                     // count=1, user D#
  g0[1] = ldsOff;                                 // lds_addr (bytes)
  g0[2] = (unsigned int)(ga & 0xFFFFFFFFull);     // global_addr[31:0]
  g0[3] = (unsigned int)((ga >> 32) & 0x1FFFFFFull) | (2u << 30); // type=2
  i32x8 g1;
  g1[0] = (int)(1u << 16);                        // data_size=1 (2 bytes)
  g1[1] = (int)(tileW << 16);                     // tensor_dim0[15:0] @63:48
  g1[2] = (int)((tileW >> 16) | (tileH << 16));   // dim0 hi | tensor_dim1 lo
  g1[3] = (int)(((tileH >> 16) & 0xFFFFu) | (tileW << 16)); // dim1 hi|tile_dim0
  g1[4] = (int)(tileH & 0xFFFFu);                 // tile_dim1; tile_dim2=0
  g1[5] = (int)rowStride;                         // tensor_dim0_stride[31:0]
  g1[6] = 0;                                      // stride hi | dim1_stride lo
  g1[7] = 0;
  i32x4 g2; g2[0] = 1; g2[1] = 0; g2[2] = 0; g2[3] = 0;  // tensor_dim2=1
  i32x4 g3; g3[0] = 0; g3[1] = 0; g3[2] = 0; g3[3] = 0;
#ifdef TDM_6ARG
  i32x8 g4; g4[0]=0; g4[1]=0; g4[2]=0; g4[3]=0; g4[4]=0; g4[5]=0; g4[6]=0; g4[7]=0;
  __builtin_amdgcn_tensor_load_to_lds(g0, g1, g2, g3, g4, 0);
#else
  __builtin_amdgcn_tensor_load_to_lds(g0, g1, g2, g3, 0);
#endif
}
#endif

// ---------------------------------------------------------------------------
// Generic batched NN GEMM: C[b] = alpha * A[b] @ B[b] + bias[b]
//   A: [M x K] f32 row-major (lda, batch stride strideA)
//   B: [K x N] f32 row-major (ldb, batch stride strideB)
//   C: f32 or bf16 (OUT_BF16), ldc, batch "stride" strideC (elements)
// Block tile 128x128, K-chunk 32; 8 waves (4M x 2N), wave tile 32x64;
// operands bf16 in LDS, v_wmma_f32_16x16x32_bf16, f32 accumulate.
// All dims are multiples of tile sizes for this problem (no bounds checks).
// ---------------------------------------------------------------------------
template <bool OUT_BF16>
__global__ __launch_bounds__(256)
void gemm_nn_wmma(const float* __restrict__ A, size_t lda, size_t strideA,
                  const float* __restrict__ B, size_t ldb, size_t strideB,
                  const float* __restrict__ bias, size_t strideBias,
                  void* __restrict__ Cv, size_t ldc, size_t strideC,
                  int Kdim, float alpha)
{
  __shared__ __align__(16) unsigned short As[128 * 32];  // [m][k]
  __shared__ __align__(16) unsigned short Bs[128 * 32];  // [n][k]

  const int tid  = threadIdx.x;
  const int lane = tid & 31;
  const int wave = tid >> 5;
  const int wr   = wave & 3;
  const int wc   = wave >> 2;
  const int g    = lane >> 4;
  const int l16  = lane & 15;

  const size_t mBase = (size_t)blockIdx.x * 128;
  const size_t nBase = (size_t)blockIdx.y * 128;
  const int batch = blockIdx.z;

  const float* Ab = A + (size_t)batch * strideA;
  const float* Bb = B + (size_t)batch * strideB;

  const v8f vz = {0.f, 0.f, 0.f, 0.f, 0.f, 0.f, 0.f, 0.f};
  v8f acc[2][4];
#pragma unroll
  for (int t = 0; t < 2; ++t)
#pragma unroll
    for (int u = 0; u < 4; ++u) acc[t][u] = vz;

  for (int k0 = 0; k0 < Kdim; k0 += 32) {
    __syncthreads();
    // ---- A tile 128x32: float4 x2 loads, pack, one ds_store_b128 / seg ----
#pragma unroll
    for (int j = 0; j < 2; ++j) {
      int s = j * 256 + tid;            // 512 segments of 8 elems
      int m = s >> 2, ks = (s & 3) * 8;
      const float4* src = (const float4*)&Ab[(mBase + m) * lda + (size_t)(k0 + ks)];
      float4 f0 = src[0], f1 = src[1];
      uint4 p;
      p.x = pk2(f0.x, f0.y); p.y = pk2(f0.z, f0.w);
      p.z = pk2(f1.x, f1.y); p.w = pk2(f1.z, f1.w);
      *(uint4*)&As[m * 32 + ks] = p;
    }
    // ---- B tile [k][n] 32x128 -> LDS [n][k] (transpose during fill) ----
#pragma unroll
    for (int j = 0; j < 4; ++j) {
      int s = j * 256 + tid;            // 1024 float4 segments
      int k = s >> 5;                   // 0..31
      int nseg = (s & 31) * 4;          // 0..124
      float4 f = *(const float4*)&Bb[(size_t)(k0 + k) * ldb + nBase + nseg];
      Bs[(nseg + 0) * 32 + k] = f2bf(f.x);
      Bs[(nseg + 1) * 32 + k] = f2bf(f.y);
      Bs[(nseg + 2) * 32 + k] = f2bf(f.z);
      Bs[(nseg + 3) * 32 + k] = f2bf(f.w);
    }
    __syncthreads();

    Frag16 fa[2];
#pragma unroll
    for (int t = 0; t < 2; ++t) {
      int row = wr * 32 + t * 16 + l16;
      fa[t].q[0] = *(const uint4*)&As[row * 32 + g * 8];
      fa[t].q[1] = *(const uint4*)&As[row * 32 + g * 8 + 16];
    }
    Frag16 fb[4];
#pragma unroll
    for (int u = 0; u < 4; ++u) {
      int col = wc * 64 + u * 16 + l16;
      fb[u].q[0] = *(const uint4*)&Bs[col * 32 + g * 8];
      fb[u].q[1] = *(const uint4*)&Bs[col * 32 + g * 8 + 16];
    }
#pragma unroll
    for (int t = 0; t < 2; ++t)
#pragma unroll
      for (int u = 0; u < 4; ++u)
        acc[t][u] = __builtin_amdgcn_wmma_f32_16x16x32_bf16(
            false, fa[t].v, false, fb[u].v, (short)0, acc[t][u], false, false);
  }

  // ---- epilogue ----
#pragma unroll
  for (int t = 0; t < 2; ++t) {
#pragma unroll
    for (int u = 0; u < 4; ++u) {
      size_t n = nBase + (size_t)(wc * 64 + u * 16 + l16);
      float bv = bias ? bias[(size_t)batch * strideBias + n] : 0.f;
#pragma unroll
      for (int r = 0; r < 8; ++r) {
        size_t m = mBase + (size_t)(wr * 32 + t * 16 + g * 8 + r);
        float val = alpha * acc[t][u][r] + bv;
        if (OUT_BF16)
          ((unsigned short*)Cv)[(size_t)batch * strideC + m * ldc + n] = f2bf(val);
        else
          ((float*)Cv)[(size_t)batch * strideC + m * ldc + n] = val;
      }
    }
  }
}

// ---------------------------------------------------------------------------
// scores[h] = (S[h] @ K[h]^T) * alpha,  S/K bf16 [rows x DH] row-major.
// Operand tiles are pure 2D bf16 sub-tile copies -> Tensor Data Mover.
// ---------------------------------------------------------------------------
__global__ __launch_bounds__(256)
void scores_wmma_tdm(const unsigned short* __restrict__ Sb,  // [H, NS, DH]
                     const unsigned short* __restrict__ Kb,  // [H, NK, DH]
                     float* __restrict__ Cout,               // [H, NS, NK]
                     float alpha)
{
  __shared__ __align__(16) unsigned short As[128 * 32];  // [m][k]
  __shared__ __align__(16) unsigned short Bs[128 * 32];  // [n][k]

  const int tid  = threadIdx.x;
  const int lane = tid & 31;
  const int wave = tid >> 5;
  const int wr   = wave & 3;
  const int wc   = wave >> 2;
  const int g    = lane >> 4;
  const int l16  = lane & 15;

  const size_t mBase = (size_t)blockIdx.x * 128;   // over NS
  const size_t nBase = (size_t)blockIdx.y * 128;   // over NK
  const int h = blockIdx.z;

  const unsigned short* Sh = Sb + (size_t)h * NS_ * DH_;
  const unsigned short* Kh = Kb + (size_t)h * NK_ * DH_;

  const v8f vz = {0.f, 0.f, 0.f, 0.f, 0.f, 0.f, 0.f, 0.f};
  v8f acc[2][4];
#pragma unroll
  for (int t = 0; t < 2; ++t)
#pragma unroll
    for (int u = 0; u < 4; ++u) acc[t][u] = vz;

#pragma unroll
  for (int k0 = 0; k0 < DH_; k0 += 32) {
    __syncthreads();
#ifdef USE_TDM
    if (tid < 32) {   // one wave drives the TDM; EXEC is ignored by tensor ops
      tdm_load_2d_bf16(&Sh[mBase * DH_ + k0],
                       (unsigned int)(size_t)(void*)As, 32u, 128u, DH_);
      tdm_load_2d_bf16(&Kh[nBase * DH_ + k0],
                       (unsigned int)(size_t)(void*)Bs, 32u, 128u, DH_);
      __builtin_amdgcn_s_wait_tensorcnt(0);
    }
#else
#pragma unroll
    for (int j = 0; j < 2; ++j) {
      int s = j * 256 + tid;
      int m = s >> 2, ks = (s & 3) * 8;
      *(uint4*)&As[m * 32 + ks] = *(const uint4*)&Sh[(mBase + m) * DH_ + k0 + ks];
      *(uint4*)&Bs[m * 32 + ks] = *(const uint4*)&Kh[(nBase + m) * DH_ + k0 + ks];
    }
#endif
    __syncthreads();

    Frag16 fa[2];
#pragma unroll
    for (int t = 0; t < 2; ++t) {
      int row = wr * 32 + t * 16 + l16;
      fa[t].q[0] = *(const uint4*)&As[row * 32 + g * 8];
      fa[t].q[1] = *(const uint4*)&As[row * 32 + g * 8 + 16];
    }
    Frag16 fb[4];
#pragma unroll
    for (int u = 0; u < 4; ++u) {
      int col = wc * 64 + u * 16 + l16;
      fb[u].q[0] = *(const uint4*)&Bs[col * 32 + g * 8];
      fb[u].q[1] = *(const uint4*)&Bs[col * 32 + g * 8 + 16];
    }
#pragma unroll
    for (int t = 0; t < 2; ++t)
#pragma unroll
      for (int u = 0; u < 4; ++u)
        acc[t][u] = __builtin_amdgcn_wmma_f32_16x16x32_bf16(
            false, fa[t].v, false, fb[u].v, (short)0, acc[t][u], false, false);
  }

  float* Ch = Cout + (size_t)h * NS_ * NK_;
#pragma unroll
  for (int t = 0; t < 2; ++t) {
#pragma unroll
    for (int u = 0; u < 4; ++u) {
      size_t n = nBase + (size_t)(wc * 64 + u * 16 + l16);
#pragma unroll
      for (int r = 0; r < 8; ++r) {
        size_t m = mBase + (size_t)(wr * 32 + t * 16 + g * 8 + r);
        Ch[m * NK_ + n] = alpha * acc[t][u][r];
      }
    }
  }
}

// ---------------------------------------------------------------------------
// Column softmax over the sentence axis (axis=1 of [H, NS, NK]), in place.
// ---------------------------------------------------------------------------
__global__ __launch_bounds__(256)
void softmax_over_rows(float* __restrict__ attn)
{
  const int col = blockIdx.x * 256 + threadIdx.x;  // 0..NK-1
  const int h   = blockIdx.y;
  float* base = attn + (size_t)h * NS_ * NK_ + col;

  float mx = -3.402823466e38f;
  float s  = 0.f;
  for (int n = 0; n < NS_; ++n) {
    float x  = base[(size_t)n * NK_];
    float mn = fmaxf(mx, x);
    s  = s * __expf(mx - mn) + __expf(x - mn);
    mx = mn;
  }
  const float inv = 1.f / s;
  for (int n = 0; n < NS_; ++n) {
    float x = base[(size_t)n * NK_];
    base[(size_t)n * NK_] = __expf(x - mx) * inv;
  }
}

// ---------------------------------------------------------------------------
// Launch: projS(bf16) -> projK(bf16) -> scores(TDM+WMMA) -> softmax -> fused
// ---------------------------------------------------------------------------
extern "C" void kernel_launch(void* const* d_in, const int* in_sizes, int n_in,
                              void* d_out, int out_size, void* d_ws, size_t ws_size,
                              hipStream_t stream)
{
  (void)in_sizes; (void)n_in; (void)out_size; (void)ws_size;

  const float* sentences = (const float*)d_in[0];  // [NS, D]
  const float* knowledge = (const float*)d_in[1];  // [NK, KDIM]
  const float* w_s       = (const float*)d_in[2];  // [H, D, DH]
  const float* b_s       = (const float*)d_in[3];  // [H, DH]
  const float* w_k       = (const float*)d_in[4];  // [H, KDIM, DH]
  const float* b_k       = (const float*)d_in[5];  // [H, DH]

  float* attnOut  = (float*)d_out;                     // [H*NS, NK]
  float* fusedOut = attnOut + (size_t)H_ * NS_ * NK_;  // [NS, H*KDIM]

  unsigned short* Sb = (unsigned short*)d_ws;          // [H, NS, DH] bf16, 4 MB
  unsigned short* Kb = Sb + (size_t)H_ * NS_ * DH_;    // [H, NK, DH] bf16, 16 MB

  const dim3 blk(256);
  const float scale = 0.088388347648318447f;           // 1/sqrt(128)

  // S[h] = bf16(sentences @ w_s[h] + b_s[h])
  gemm_nn_wmma<true><<<dim3(NS_ / 128, DH_ / 128, H_), blk, 0, stream>>>(
      sentences, (size_t)D_, (size_t)0,
      w_s, (size_t)DH_, (size_t)D_ * DH_,
      b_s, (size_t)DH_,
      (void*)Sb, (size_t)DH_, (size_t)NS_ * DH_,
      D_, 1.0f);

  // K[h] = bf16(knowledge @ w_k[h] + b_k[h])
  gemm_nn_wmma<true><<<dim3(NK_ / 128, DH_ / 128, H_), blk, 0, stream>>>(
      knowledge, (size_t)KDIM_, (size_t)0,
      w_k, (size_t)DH_, (size_t)KDIM_ * DH_,
      b_k, (size_t)DH_,
      (void*)Kb, (size_t)DH_, (size_t)NK_ * DH_,
      KDIM_, 1.0f);

  // scores -> attn output region (pre-softmax)
  scores_wmma_tdm<<<dim3(NS_ / 128, NK_ / 128, H_), blk, 0, stream>>>(
      Sb, Kb, attnOut, scale);

  // softmax over sentence axis, in place
  softmax_over_rows<<<dim3(NK_ / 256, H_), blk, 0, stream>>>(attnOut);

  // fused[:, h*KDIM : (h+1)*KDIM] = attn[h] @ knowledge
  gemm_nn_wmma<false><<<dim3(NS_ / 128, KDIM_ / 128, H_), blk, 0, stream>>>(
      attnOut, (size_t)NK_, (size_t)NS_ * NK_,
      knowledge, (size_t)KDIM_, (size_t)0,
      nullptr, (size_t)0,
      (void*)fusedOut, (size_t)(H_ * KDIM_), (size_t)KDIM_,
      NK_, 1.0f);
}
